// RecurrentEvolutionGNN_76836964926334
// MI455X (gfx1250) — compile-verified, matrix operand order checked by reading, not run on hardware
//
#include <hip/hip_runtime.h>
#include <math.h>

#define NN 10000
#define EE 40000
#define DIMD 256
#define HEADS_ 8
#define KVH_ 2
#define HD_ 32
#define WEVO 8
#define THD_ 64

typedef __attribute__((ext_vector_type(16))) __bf16 v16bf;
typedef __attribute__((ext_vector_type(8))) float v8f;

// Native f32->bf16 (fptrunc, RNE). Lowers to v_cvt_pk_bf16_f32-class ops on
// gfx1250; always legal in clang for AMDGPU so no backend-selection risk.
__device__ __forceinline__ __bf16 f2bf(float f) { return (__bf16)f; }

// pack two floats into two consecutive bf16 (low = a, high = b)
__device__ __forceinline__ unsigned pk2bf(float a, float b) {
  union { __bf16 h[2]; unsigned u; } p;
  p.h[0] = (__bf16)a; p.h[1] = (__bf16)b;
  return p.u;
}

__device__ __forceinline__ float wave_sum(float v) {
  #pragma unroll
  for (int off = 16; off > 0; off >>= 1) v += __shfl_xor(v, off, 32);
  return v;
}

__device__ __forceinline__ unsigned f2ord(float f) {
  unsigned u = __float_as_uint(f);
  return (u & 0x80000000u) ? ~u : (u | 0x80000000u);
}
__device__ __forceinline__ float ord2f(unsigned u) {
  unsigned v = (u & 0x80000000u) ? (u & 0x7FFFFFFFu) : ~u;
  return __uint_as_float(v);
}

__device__ __forceinline__ float gelu_exact(float v) {
  return 0.5f * v * (1.f + erff(v * 0.70710678118654752f));
}

// ---------------------------------------------------------------------------
// C[M,Nd] = act(A[M,K] @ B[K,Nd] + bias[Nd]) with bf16 WMMA.
// 128x64 block tile, 8 waves (4x2), each wave owns 32x32 ->
// 4 x v_wmma_f32_16x16x32_bf16 per k-step from 8 ds_load_b128.
// Double-buffered LDS, software pipelined (global loads issued before the
// WMMAs, packed-bf16 stores after), one barrier per k-step, L2 prefetch of
// the k+2 tiles. Block-uniform fast/slow paths keep loads unpredicated.
// ---------------------------------------------------------------------------
#define BM 128
#define BN 64
#define BK 32
#define KST 40   // bf16 units per LDS row; 80B stride (16B aligned)

__global__ __launch_bounds__(256) void gemm_wmma_kernel(
    const float* __restrict__ A, const float* __restrict__ B,
    const float* __restrict__ bias, float* __restrict__ C,
    int M, int Nd, int K, int act) {
  __shared__ __align__(16) __bf16 As[2][BM * KST];
  __shared__ __align__(16) __bf16 Bs[2][BN * KST];
  const int t = threadIdx.x;
  const int lane = t & 31, wave = t >> 5;
  const int wm = wave >> 1, wn = wave & 1;    // 4x2 wave grid
  const int bm0 = blockIdx.y * BM;
  const int bn0 = blockIdx.x * BN;

  v8f acc[2][2];
  #pragma unroll
  for (int i = 0; i < 2; i++)
    #pragma unroll
    for (int j = 0; j < 2; j++)
      acc[i][j] = (v8f){0.f, 0.f, 0.f, 0.f, 0.f, 0.f, 0.f, 0.f};

  // staging maps
  const int arow = t >> 1;            // A: 128 rows, 2 threads/row
  const int akb  = (t & 1) << 4;      //    16 consecutive k each
  const int bkq  = (t >> 4) << 1;     // B: k pair base (0,2,...,30)
  const int bnb  = (t & 15) << 2;     //    4 consecutive n each

  const bool mfull = (bm0 + BM <= M);     // block-uniform
  const bool nfull = (bn0 + BN <= Nd);    // block-uniform

  float areg[16];
  float breg0[4], breg1[4];

  auto loadA = [&](int k0) {
    if (mfull) {
      const float* ap = A + (size_t)(bm0 + arow) * K + k0 + akb;
      #pragma unroll
      for (int i = 0; i < 16; i += 4) {
        float4 v = *(const float4*)(ap + i);
        areg[i] = v.x; areg[i + 1] = v.y; areg[i + 2] = v.z; areg[i + 3] = v.w;
      }
    } else {
      int gr = bm0 + arow;
      if (gr < M) {
        const float* ap = A + (size_t)gr * K + k0 + akb;
        #pragma unroll
        for (int i = 0; i < 16; i++) areg[i] = ap[i];
      } else {
        #pragma unroll
        for (int i = 0; i < 16; i++) areg[i] = 0.f;
      }
    }
  };
  auto loadB = [&](int k0) {
    const float* bp0 = B + (size_t)(k0 + bkq) * Nd + bn0 + bnb;
    const float* bp1 = bp0 + Nd;
    if (nfull) {
      float4 v0 = *(const float4*)bp0;
      float4 v1 = *(const float4*)bp1;
      breg0[0] = v0.x; breg0[1] = v0.y; breg0[2] = v0.z; breg0[3] = v0.w;
      breg1[0] = v1.x; breg1[1] = v1.y; breg1[2] = v1.z; breg1[3] = v1.w;
    } else {
      #pragma unroll
      for (int i = 0; i < 4; i++) {
        int n = bn0 + bnb + i;
        breg0[i] = (n < Nd) ? bp0[i] : 0.f;
        breg1[i] = (n < Nd) ? bp1[i] : 0.f;
      }
    }
  };
  auto stageA = [&](int buf) {
    union { unsigned u[4]; uint4 q; } p0, p1;
    #pragma unroll
    for (int i = 0; i < 4; i++) {
      p0.u[i] = pk2bf(areg[2 * i], areg[2 * i + 1]);
      p1.u[i] = pk2bf(areg[8 + 2 * i], areg[8 + 2 * i + 1]);
    }
    *(uint4*)(&As[buf][arow * KST + akb]) = p0.q;
    *(uint4*)(&As[buf][arow * KST + akb + 8]) = p1.q;
  };
  auto stageB = [&](int buf) {
    #pragma unroll
    for (int i = 0; i < 4; i++)  // Bs[n][bkq..bkq+1] as one b32 store
      *(unsigned*)(&Bs[buf][(bnb + i) * KST + bkq]) = pk2bf(breg0[i], breg1[i]);
  };
  auto compute = [&](int buf) {
    const int half = lane >> 4, l15 = lane & 15;
    union { v16bf v; uint4 q[2]; } af[2], bf[2];
    #pragma unroll
    for (int i = 0; i < 2; i++) {
      const __bf16* ap = &As[buf][(wm * 32 + i * 16 + l15) * KST];
      af[i].q[0] = *(const uint4*)(ap + half * 8);
      af[i].q[1] = *(const uint4*)(ap + 16 + half * 8);
    }
    #pragma unroll
    for (int j = 0; j < 2; j++) {
      const __bf16* bp = &Bs[buf][(wn * 32 + j * 16 + l15) * KST];
      bf[j].q[0] = *(const uint4*)(bp + half * 8);
      bf[j].q[1] = *(const uint4*)(bp + 16 + half * 8);
    }
    #pragma unroll
    for (int i = 0; i < 2; i++)
      #pragma unroll
      for (int j = 0; j < 2; j++)
        acc[i][j] = __builtin_amdgcn_wmma_f32_16x16x32_bf16(
            false, af[i].v, false, bf[j].v, (short)0, acc[i][j], false, false);
  };

  const int nk = K / BK;
  loadA(0); loadB(0);
  stageA(0); stageB(0);
  __syncthreads();

  for (int kt = 0; kt < nk; kt++) {
    const int cur = kt & 1;
    const bool hasNext = (kt + 1 < nk);
    if (hasNext) { loadA((kt + 1) * BK); loadB((kt + 1) * BK); }
    if (kt + 2 < nk) {   // warm L2 for the k+2 tiles (global_prefetch_b8)
      if (mfull || (bm0 + arow) < M)
        __builtin_prefetch(A + (size_t)(bm0 + arow) * K + (kt + 2) * BK + akb, 0, 1);
      __builtin_prefetch(B + (size_t)((kt + 2) * BK + bkq) * Nd + bn0 + bnb, 0, 1);
    }
    compute(cur);
    if (hasNext) {
      stageA(1 - cur); stageB(1 - cur);
      __syncthreads();
    }
  }

  // C/D layout: VGPR r -> M = r (lanes 0-15) or r+8 (lanes 16-31), N = lane&15
  const int l15 = lane & 15;
  const int rbase = (lane < 16) ? 0 : 8;
  #pragma unroll
  for (int i = 0; i < 2; i++) {
    #pragma unroll
    for (int j = 0; j < 2; j++) {
      int col = bn0 + wn * 32 + j * 16 + l15;
      if (col < Nd) {
        float bv = bias ? bias[col] : 0.f;
        #pragma unroll
        for (int r = 0; r < 8; r++) {
          int row = bm0 + wm * 32 + i * 16 + rbase + r;
          if (row < M) {
            float v = acc[i][j][r] + bv;
            if (act == 1) v = gelu_exact(v);
            C[(size_t)row * Nd + col] = v;
          }
        }
      }
    }
  }
}

// ---------------------------------------------------------------------------
// Small helper kernels
// ---------------------------------------------------------------------------
__global__ void fill_kernel(unsigned* p, unsigned v, int count) {
  int i = blockIdx.x * blockDim.x + threadIdx.x;
  if (i < count) p[i] = v;
}

// one wave per row: out[r] = f(dot(A[r,:], w) + b)
// mode 0: none, 1: sigmoid, 2: conf = 1/(1+max(sqrt(mean),1e-6)) (w,b unused)
__global__ __launch_bounds__(256) void rowdot_kernel(
    const float* __restrict__ A, const float* __restrict__ w,
    const float* __restrict__ b, float* __restrict__ out,
    int rows, int K, int mode) {
  int wave = threadIdx.x >> 5, lane = threadIdx.x & 31;
  int r = blockIdx.x * 8 + wave;
  if (r >= rows) return;
  float acc = 0.f;
  for (int k = lane; k < K; k += 32)
    acc += A[(size_t)r * K + k] * (w ? w[k] : 1.f);
  acc = wave_sum(acc);
  if (lane == 0) {
    float v;
    if (mode == 2) {
      v = 1.f / (1.f + fmaxf(sqrtf(acc / (float)K), 1e-6f));
    } else {
      v = acc + (b ? b[0] : 0.f);
      if (mode == 1) v = 1.f / (1.f + expf(-v));
    }
    out[r] = v;
  }
}

__global__ void rope_kernel(float* __restrict__ X, const float* __restrict__ t,
                            int nh, int count) {
  int idx = blockIdx.x * blockDim.x + threadIdx.x;
  if (idx >= count) return;                 // count = N*nh*16
  int p = idx & 15;
  int h = (idx >> 4) % nh;
  int n = idx / (16 * nh);
  float invf = powf(10000.f, -((float)(2 * p)) / (float)HD_);
  float fr = t[n] * invf;
  float c = cosf(fr), s = sinf(fr);
  float* base = X + ((size_t)n * nh + h) * HD_ + 2 * p;
  float x1 = base[0], x2 = base[1];
  base[0] = x1 * c - x2 * s;
  base[1] = x2 * c + x1 * s;
}

__global__ void temp_kernel(const float* __restrict__ ltb,
                            const float* __restrict__ nburst,
                            const float* __restrict__ Wbt,
                            const float* __restrict__ bbt,
                            float* __restrict__ temp) {
  int idx = blockIdx.x * blockDim.x + threadIdx.x;
  if (idx >= NN * HEADS_) return;
  int n = idx >> 3, h = idx & 7;
  temp[idx] = expf(ltb[h] + nburst[n] * Wbt[h] + bbt[h]);
}

// one wave per edge: trust-attention + fused Wt_out logit
__global__ __launch_bounds__(256) void trust_kernel(
    const float* __restrict__ tQ, const float* __restrict__ tK,
    const float* __restrict__ tV, const float* __restrict__ unc,
    const float* __restrict__ conf, const float* __restrict__ evo_times,
    const float* __restrict__ nlt, const float* __restrict__ time_decay,
    const float* __restrict__ Wt_out, const float* __restrict__ bt_out,
    const int* __restrict__ src, const int* __restrict__ dst,
    float* __restrict__ trust) {
  int wave = threadIdx.x >> 5, lane = threadIdx.x & 31;
  int e = blockIdx.x * 8 + wave;
  if (e >= EE) return;
  int sn = src[e], dn = dst[e];
  int n = lane >> 3, w = lane & 7;            // head 0..3, window 0..7
  const float* q  = tQ + (size_t)dn * DIMD + n * THD_;
  const float* kk = tK + ((size_t)sn * WEVO + w) * DIMD + n * THD_;
  float acc = 0.f;
  #pragma unroll 8
  for (int j = 0; j < THD_; j++) acc += q[j] * kk[j];
  float a = acc * 0.125f;                      // THD^-0.5
  float dt = fmaxf(nlt[dn] - evo_times[sn * WEVO + w], 0.f);
  a -= fabsf(time_decay[0]) * dt;
  a *= unc[sn * WEVO + w];
  float mx = a;
  #pragma unroll
  for (int off = 1; off < 8; off <<= 1) mx = fmaxf(mx, __shfl_xor(mx, off, 32));
  float ex = expf(a - mx);
  float sm = ex;
  #pragma unroll
  for (int off = 1; off < 8; off <<= 1) sm += __shfl_xor(sm, off, 32);
  float al = ex / sm;
  const float* vv = tV + ((size_t)sn * WEVO + w) * DIMD + n * THD_;
  const float* wt = Wt_out + n * THD_;
  float lp = 0.f;
  #pragma unroll 8
  for (int j = 0; j < THD_; j++) lp += vv[j] * wt[j];
  lp *= al;
  lp = wave_sum(lp);                           // sum over n,w,d = logit
  if (lane == 0) {
    float lg = (lp + bt_out[0]) * conf[dn] * conf[sn];
    trust[e] = 1.f / (1.f + expf(-lg));
  }
}

__global__ void edge_attn_kernel(const float* __restrict__ Qr,
                                 const float* __restrict__ Kr,
                                 const float* __restrict__ ek,
                                 const float* __restrict__ rec,
                                 const float* __restrict__ temp,
                                 const float* __restrict__ trust,
                                 const int* __restrict__ src,
                                 const int* __restrict__ dst,
                                 float* __restrict__ attn) {
  int idx = blockIdx.x * blockDim.x + threadIdx.x;
  if (idx >= EE * HEADS_) return;
  int e = idx >> 3, h = idx & 7;
  int sn = src[e], dn = dst[e];
  int kvh = h >> 2;
  const float* q  = Qr + (size_t)dn * DIMD + h * HD_;
  const float* k  = Kr + (size_t)sn * (KVH_ * HD_) + kvh * HD_;
  const float* ep = ek + (size_t)e * (KVH_ * HD_) + kvh * HD_;
  float acc = 0.f;
  #pragma unroll
  for (int d = 0; d < HD_; d++) acc += q[d] * (k[d] + ep[d]);
  float a = acc * 0.17677669529663687f;        // HD^-0.5
  a = a / fmaxf(temp[dn * HEADS_ + h], 0.01f);
  a += rec[(size_t)e * HEADS_ + h];
  a *= trust[e];
  attn[idx] = a;
}

__global__ void segmax_kernel(const float* __restrict__ attn,
                              const int* __restrict__ dst,
                              unsigned* __restrict__ mseg) {
  int idx = blockIdx.x * blockDim.x + threadIdx.x;
  if (idx >= EE * HEADS_) return;
  int e = idx >> 3, h = idx & 7;
  atomicMax(&mseg[dst[e] * HEADS_ + h], f2ord(attn[idx]));
}

__global__ void exseg_kernel(float* __restrict__ attn,
                             const int* __restrict__ dst,
                             const unsigned* __restrict__ mseg,
                             float* __restrict__ sseg) {
  int idx = blockIdx.x * blockDim.x + threadIdx.x;
  if (idx >= EE * HEADS_) return;
  int e = idx >> 3, h = idx & 7;
  float ex = expf(attn[idx] - ord2f(mseg[dst[e] * HEADS_ + h]));
  attn[idx] = ex;
  atomicAdd(&sseg[dst[e] * HEADS_ + h], ex);
}

__global__ void msg_kernel(const float* __restrict__ attn,
                           const float* __restrict__ Vv,
                           const int* __restrict__ src,
                           const int* __restrict__ dst,
                           const float* __restrict__ sseg,
                           float* __restrict__ agg) {
  int idx = blockIdx.x * blockDim.x + threadIdx.x;
  if (idx >= EE * HEADS_) return;
  int e = idx >> 3, h = idx & 7;
  int sn = src[e], dn = dst[e];
  float alpha = attn[idx] / (sseg[dn * HEADS_ + h] + 1e-16f);
  const float* vp = Vv + (size_t)sn * (KVH_ * HD_) + (h >> 2) * HD_;
  float* ap = agg + (size_t)dn * DIMD + h * HD_;
  #pragma unroll
  for (int d = 0; d < HD_; d++) atomicAdd(&ap[d], alpha * vp[d]);
}

__global__ void tsum_kernel(const float* __restrict__ time_emb,
                            const int* __restrict__ src,
                            float* __restrict__ tsum) {
  int idx = blockIdx.x * blockDim.x + threadIdx.x;
  if (idx >= EE * DIMD) return;
  int e = idx >> 8, d = idx & 255;
  atomicAdd(&tsum[(size_t)src[e] * DIMD + d], time_emb[idx]);
}

__global__ void cnt_kernel(const int* __restrict__ src, float* __restrict__ cnt) {
  int e = blockIdx.x * blockDim.x + threadIdx.x;
  if (e < EE) atomicAdd(&cnt[src[e]], 1.f);
}

__global__ void fusedcopy_kernel(const float* __restrict__ agg,
                                 const float* __restrict__ memh,
                                 float* __restrict__ fused) {
  int idx = blockIdx.x * blockDim.x + threadIdx.x;
  if (idx >= NN * 2 * DIMD) return;
  int n = idx >> 9, d = idx & 511;
  fused[idx] = (d < DIMD) ? agg[n * DIMD + d] : memh[n * DIMD + (d - DIMD)];
}

__global__ void topk_kernel(const float* __restrict__ logits,
                            float* __restrict__ wexp) {
  int n = blockIdx.x * blockDim.x + threadIdx.x;
  if (n >= NN) return;
  float l[4];
  #pragma unroll
  for (int j = 0; j < 4; j++) l[j] = logits[n * 4 + j];
  int i0 = 0; float v0 = l[0];
  #pragma unroll
  for (int j = 1; j < 4; j++) if (l[j] > v0) { v0 = l[j]; i0 = j; }
  int i1 = -1; float v1 = -1e30f;
  #pragma unroll
  for (int j = 0; j < 4; j++) if (j != i0 && l[j] > v1) { v1 = l[j]; i1 = j; }
  float e1 = expf(v1 - v0);
  float s = 1.f + e1;
  float w[4] = {0.f, 0.f, 0.f, 0.f};
  w[i0] = 1.f / s;
  w[i1] = e1 / s;
  #pragma unroll
  for (int j = 0; j < 4; j++) wexp[n * 4 + j] = w[j];
}

__global__ void moeacc_kernel(float* __restrict__ moe,
                              const float* __restrict__ eo,
                              const float* __restrict__ wexp, int e) {
  int idx = blockIdx.x * blockDim.x + threadIdx.x;
  if (idx >= NN * DIMD) return;
  int n = idx >> 8;
  moe[idx] += wexp[n * 4 + e] * eo[idx];
}

__global__ void stc_kernel(const float* __restrict__ tsum,
                           const float* __restrict__ cnt,
                           float* __restrict__ stc) {
  int idx = blockIdx.x * blockDim.x + threadIdx.x;
  if (idx >= NN * DIMD) return;
  int n = idx >> 8;
  float tc = tsum[idx] / (cnt[n] + 1e-6f);
  stc[idx] = tc / (1.f + expf(-tc));           // silu
}

// one wave per row: adaptive LN + residual gate
__global__ __launch_bounds__(256) void final_kernel(
    const float* __restrict__ aggo, const float* __restrict__ ss,
    const float* __restrict__ beta, const float* __restrict__ x,
    float* __restrict__ out) {
  int wave = threadIdx.x >> 5, lane = threadIdx.x & 31;
  int row = blockIdx.x * 8 + wave;
  if (row >= NN) return;
  float v[8], s = 0.f, sq = 0.f;
  #pragma unroll
  for (int i = 0; i < 8; i++) {
    v[i] = aggo[(size_t)row * DIMD + lane + 32 * i];
    s += v[i]; sq += v[i] * v[i];
  }
  s = wave_sum(s); sq = wave_sum(sq);
  float mu = s / 256.f;
  float var = sq / 256.f - mu * mu;
  float rstd = rsqrtf(var + 1e-5f);
  float b = beta[row];
  #pragma unroll
  for (int i = 0; i < 8; i++) {
    int d = lane + 32 * i;
    float nm = (v[i] - mu) * rstd;
    float a = nm * (1.f + ss[(size_t)row * 512 + d]) + ss[(size_t)row * 512 + 256 + d];
    out[(size_t)row * DIMD + d] = b * x[(size_t)row * DIMD + d] + (1.f - b) * a;
  }
}

// ---------------------------------------------------------------------------
extern "C" void kernel_launch(void* const* d_in, const int* in_sizes, int n_in,
                              void* d_out, int out_size, void* d_ws, size_t ws_size,
                              hipStream_t stream) {
  (void)in_sizes; (void)n_in; (void)out_size; (void)ws_size;
  // inputs
  const float* x          = (const float*)d_in[0];
  const int*   eidx       = (const int*)d_in[1];
  const int*   src        = eidx;
  const int*   dst        = eidx + EE;
  const float* edge_emb   = (const float*)d_in[2];
  const float* time_emb   = (const float*)d_in[3];
  const float* nlt        = (const float*)d_in[4];
  const float* nburst     = (const float*)d_in[5];
  const float* mem_vals   = (const float*)d_in[6];
  const float* mem_vars   = (const float*)d_in[7];
  const float* mem_hidden = (const float*)d_in[8];
  const float* evo_vals   = (const float*)d_in[9];
  const float* evo_times  = (const float*)d_in[10];
  // params (dict order)
  const float* Wq  = (const float*)d_in[11];  const float* bq  = (const float*)d_in[12];
  const float* Wk  = (const float*)d_in[13];  const float* bk  = (const float*)d_in[14];
  const float* Wv  = (const float*)d_in[15];  const float* bv  = (const float*)d_in[16];
  const float* Wo  = (const float*)d_in[17];  const float* bo  = (const float*)d_in[18];
  const float* We  = (const float*)d_in[19];  const float* be  = (const float*)d_in[20];
  const float* ltb = (const float*)d_in[21];
  const float* Wbt = (const float*)d_in[22];  const float* bbt = (const float*)d_in[23];
  const float* tWq = (const float*)d_in[24];  const float* tbq = (const float*)d_in[25];
  const float* tWk = (const float*)d_in[26];  const float* tbk = (const float*)d_in[27];
  const float* tWv = (const float*)d_in[28];  const float* tbv = (const float*)d_in[29];
  const float* tdcy= (const float*)d_in[30];
  const float* Wu  = (const float*)d_in[31];  const float* bu  = (const float*)d_in[32];
  const float* Wt_out = (const float*)d_in[33]; const float* bt_out = (const float*)d_in[34];
  const float* Wrf1= (const float*)d_in[35];  const float* brf1= (const float*)d_in[36];
  const float* Wrf2= (const float*)d_in[37];  const float* brf2= (const float*)d_in[38];
  const float* Wrb = (const float*)d_in[39];  const float* brb = (const float*)d_in[40];
  const float* Wg  = (const float*)d_in[41];  const float* bg  = (const float*)d_in[42];
  const float* W1  = (const float*)d_in[43];  const float* b1  = (const float*)d_in[44];
  const float* W2  = (const float*)d_in[45];  const float* b2  = (const float*)d_in[46];
  const float* Wan = (const float*)d_in[47];  const float* ban = (const float*)d_in[48];
  const float* Wgl = (const float*)d_in[49];  const float* bgl = (const float*)d_in[50];

  float* out = (float*)d_out;

  // workspace layout (floats)
  float* ws = (float*)d_ws;
  size_t off = 0;
  auto WSA = [&](size_t n) { float* p = ws + off; off += n; return p; };
  float* Qr     = WSA((size_t)NN * 256);
  float* Kr     = WSA((size_t)NN * 64);
  float* Vv     = WSA((size_t)NN * 64);
  float* ek     = WSA((size_t)EE * 64);
  float* rec    = WSA((size_t)EE * 8);
  float* temp   = WSA((size_t)NN * 8);
  float* tQ     = WSA((size_t)NN * 256);
  float* tK     = WSA((size_t)NN * 8 * 256);
  float* tV     = WSA((size_t)NN * 8 * 256);
  float* uncb   = WSA((size_t)NN * 8);
  float* confb  = WSA((size_t)NN);
  float* trustb = WSA((size_t)EE);
  float* attnb  = WSA((size_t)EE * 8);
  unsigned* mseg = (unsigned*)WSA((size_t)NN * 8);
  float* sseg   = WSA((size_t)NN * 8);
  float* agg1   = WSA((size_t)NN * 256);
  float* tsum   = WSA((size_t)NN * 256);
  float* cntb   = WSA((size_t)NN);
  float* betab  = WSA((size_t)NN);
  float* logitsb= WSA((size_t)NN * 4);
  float* wexpb  = WSA((size_t)NN * 4);
  // reuse pool over tK/tV (only needed until trust_kernel completes):
  float* fused = tK;                           // N*512
  float* t1    = fused + (size_t)NN * 512;     // N*256
  float* agg2  = t1 + (size_t)NN * 256;        // N*256
  float* hbuf  = agg2 + (size_t)NN * 256;      // N*512
  float* eob   = hbuf + (size_t)NN * 512;      // N*256
  float* moe   = eob + (size_t)NN * 256;       // N*256
  float* aggo  = moe + (size_t)NN * 256;       // N*256
  float* stc   = aggo + (size_t)NN * 256;      // N*256
  float* ssb   = stc + (size_t)NN * 256;       // N*512   (3072/node < 4096/node pool)

  auto gemm = [&](const float* Ap, const float* Bp, const float* biasp, float* Cp,
                  int M, int Nd, int K, int act) {
    dim3 g((Nd + BN - 1) / BN, (M + BM - 1) / BM);
    gemm_wmma_kernel<<<g, dim3(256), 0, stream>>>(Ap, Bp, biasp, Cp, M, Nd, K, act);
  };
  auto ew = [&](int count) { return dim3((count + 255) / 256); };

  // init accumulators
  fill_kernel<<<ew(NN * 8), 256, 0, stream>>>(mseg, 0x007FFFFFu, NN * 8); // ord(-inf)
  fill_kernel<<<ew(NN * 8), 256, 0, stream>>>((unsigned*)sseg, 0u, NN * 8);
  fill_kernel<<<ew(NN * 256), 256, 0, stream>>>((unsigned*)agg1, 0u, NN * 256);
  fill_kernel<<<ew(NN * 256), 256, 0, stream>>>((unsigned*)tsum, 0u, NN * 256);
  fill_kernel<<<ew(NN), 256, 0, stream>>>((unsigned*)cntb, 0u, NN);

  // projections (WMMA GEMMs)
  gemm(x, Wq, bq, Qr, NN, 256, 256, 0);
  gemm(x, Wk, bk, Kr, NN, 64, 256, 0);
  gemm(x, Wv, bv, Vv, NN, 64, 256, 0);
  gemm(edge_emb, We, be, ek, EE, 64, 256, 0);
  gemm(time_emb, Wrb, brb, rec, EE, 8, 256, 0);
  gemm(mem_vals, tWq, tbq, tQ, NN, 256, 256, 0);
  gemm(evo_vals, tWk, tbk, tK, NN * WEVO, 256, 256, 0);  // per-node (gather commutes)
  gemm(evo_vals, tWv, tbv, tV, NN * WEVO, 256, 256, 0);

  // rope, temperature, uncertainty, confidence
  rope_kernel<<<ew(NN * HEADS_ * 16), 256, 0, stream>>>(Qr, nlt, HEADS_, NN * HEADS_ * 16);
  rope_kernel<<<ew(NN * KVH_ * 16), 256, 0, stream>>>(Kr, nlt, KVH_, NN * KVH_ * 16);
  temp_kernel<<<ew(NN * 8), 256, 0, stream>>>(ltb, nburst, Wbt, bbt, temp);
  rowdot_kernel<<<dim3((NN * WEVO + 7) / 8), 256, 0, stream>>>(
      evo_vals, Wu, bu, uncb, NN * WEVO, 256, 1);
  rowdot_kernel<<<dim3((NN + 7) / 8), 256, 0, stream>>>(
      mem_vars, nullptr, nullptr, confb, NN, 256, 2);

  // trust attention (fused pooled @ Wt_out)
  trust_kernel<<<dim3((EE + 7) / 8), 256, 0, stream>>>(
      tQ, tK, tV, uncb, confb, evo_times, nlt, tdcy, Wt_out, bt_out, src, dst, trustb);

  // edge attention + segment softmax + aggregation
  edge_attn_kernel<<<ew(EE * 8), 256, 0, stream>>>(Qr, Kr, ek, rec, temp, trustb,
                                                   src, dst, attnb);
  segmax_kernel<<<ew(EE * 8), 256, 0, stream>>>(attnb, dst, mseg);
  exseg_kernel<<<ew(EE * 8), 256, 0, stream>>>(attnb, dst, mseg, sseg);
  msg_kernel<<<ew(EE * 8), 256, 0, stream>>>(attnb, Vv, src, dst, sseg, agg1);

  // time pooling over src
  tsum_kernel<<<ew(EE * 256), 256, 0, stream>>>(time_emb, src, tsum);
  cnt_kernel<<<ew(EE), 256, 0, stream>>>(src, cntb);

  // refine MLP (pool region now free)
  fusedcopy_kernel<<<ew(NN * 512), 256, 0, stream>>>(agg1, mem_hidden, fused);
  gemm(fused, Wrf1, brf1, t1, NN, 256, 512, 1);   // gelu
  gemm(t1, Wrf2, brf2, agg2, NN, 256, 256, 0);

  // MoE
  gemm(agg2, Wg, bg, logitsb, NN, 4, 256, 0);
  topk_kernel<<<ew(NN), 256, 0, stream>>>(logitsb, wexpb);
  fill_kernel<<<ew(NN * 256), 256, 0, stream>>>((unsigned*)moe, 0u, NN * 256);
  for (int e = 0; e < 4; e++) {
    gemm(agg2, W1 + (size_t)e * 256 * 512, b1 + (size_t)e * 512, hbuf, NN, 512, 256, 1);
    gemm(hbuf, W2 + (size_t)e * 512 * 256, b2 + (size_t)e * 256, eob, NN, 256, 512, 0);
    moeacc_kernel<<<ew(NN * 256), 256, 0, stream>>>(moe, eob, wexpb, e);
  }
  gemm(moe, Wo, bo, aggo, NN, 256, 256, 0);

  // adaptive LN conditioning
  stc_kernel<<<ew(NN * 256), 256, 0, stream>>>(tsum, cntb, stc);
  gemm(stc, Wan, ban, ssb, NN, 512, 256, 0);
  rowdot_kernel<<<dim3((NN + 7) / 8), 256, 0, stream>>>(x, Wgl, bgl, betab, NN, 256, 1);

  final_kernel<<<dim3((NN + 7) / 8), 256, 0, stream>>>(aggo, ssb, betab, x, out);
}
